// AnalogRNNModel_59665685676355
// MI455X (gfx1250) — matrix operand order repaired and checked by reading, not compiled, not run on hardware
//
#include <hip/hip_runtime.h>
#include <hip/hip_bf16.h>

// ---------------------------------------------------------------------------
// AnalogRNN forward for MI455X (gfx1250), wave32 + WMMA bf16.
//
// Pipeline (all on `stream`):
//   1. ctx_kernel       : quantized context MLP -> cbase[b][h] (32x256 f32)
//   2. rec_kernel<1>    : fused audio projection + recurrence -> out_buf bf16
//   3. proj_kernel(L1)  : out_buf @ w_ih1.T + bias -> pre_buf f32   (WMMA)
//   4. rec_kernel<0>    : recurrence reading pre_buf -> out_buf
//   5. proj_kernel(L2) / rec_kernel<0>(L2)
//   6. head_kernel      : gelu(out @ q(Wh1).T + b) @ q(Wh2).T + b -> d_out
// ---------------------------------------------------------------------------

typedef __attribute__((ext_vector_type(16))) __bf16 v16bf;
typedef __attribute__((ext_vector_type(8)))  __bf16 v8bf;
typedef __attribute__((ext_vector_type(8)))  float  v8f;

#define TSEQ 8192
#define BATCH 32
#define HID 256
#define FIN 10
#define HPAD 264   // LDS row pad: 264 halves = 528B -> 4-bank rotation per row

// fake_quantize forward: round((clip(w,-1,1)+1)*15.5)/15.5 - 1  (RNE like jnp.round)
__device__ __forceinline__ float fq(float w) {
    float wc = fminf(fmaxf(w, -1.0f), 1.0f);
    return rintf((wc + 1.0f) * 15.5f) * (1.0f / 15.5f) - 1.0f;
}

// hardware v_tanh_f32 when available (gfx1250 trans op), else ocml tanh
__device__ __forceinline__ float fast_tanh(float x) {
#if __has_builtin(__builtin_amdgcn_tanhf)
    return __builtin_amdgcn_tanhf(x);
#elif __has_builtin(__builtin_amdgcn_tanh_f32)
    return __builtin_amdgcn_tanh_f32(x);
#else
    return tanhf(x);
#endif
}

__device__ __forceinline__ v8f wmma_bf16(v16bf a, v16bf b, v8f c) {
    return __builtin_amdgcn_wmma_f32_16x16x32_bf16(
        /*neg_a=*/false, a, /*neg_b=*/false, b,
        /*c_mod=*/(short)0, c, /*reuse_a=*/false, /*reuse_b=*/false);
}

// ---------------------------------------------------------------------------
// 1) Context MLP: raw_ctx(32x9) -> relu(q(W1)) -> tanh(q(W2)) -> cbase(32x256)
//    cbase[b][h] = sum_c ctx[b][c]*w_ih0[h][1+c] + b_ih0[h] + b_hh0[h]
// ---------------------------------------------------------------------------
__global__ __launch_bounds__(256)
void ctx_kernel(const float* __restrict__ x,
                const float* __restrict__ W1, const float* __restrict__ b1,
                const float* __restrict__ W2, const float* __restrict__ b2,
                const float* __restrict__ w_ih0,
                const float* __restrict__ b_ih0, const float* __restrict__ b_hh0,
                float* __restrict__ cbase)
{
    __shared__ float raw[BATCH][12];
    __shared__ float h1[BATCH][64];
    __shared__ float ctx[BATCH][32];
    int tid = threadIdx.x;

    for (int i = tid; i < BATCH * 9; i += 256) {
        int b = i / 9, j = i % 9;
        raw[b][j] = x[(long)b * TSEQ * FIN + 1 + j];   // x[b][0][1+j]
    }
    __syncthreads();

    for (int i = tid; i < BATCH * 64; i += 256) {
        int b = i / 64, o = i % 64;
        float s = b1[o];
        #pragma unroll
        for (int j = 0; j < 9; ++j) s += raw[b][j] * fq(W1[o * 9 + j]);
        h1[b][o] = fmaxf(s, 0.0f);
    }
    __syncthreads();

    for (int i = tid; i < BATCH * 32; i += 256) {
        int b = i / 32, c = i % 32;
        float s = b2[c];
        #pragma unroll 8
        for (int o = 0; o < 64; ++o) s += h1[b][o] * fq(W2[c * 64 + o]);
        ctx[b][c] = tanhf(s);
    }
    __syncthreads();

    for (int i = tid; i < BATCH * HID; i += 256) {
        int b = i / HID, h = i % HID;
        float s = b_ih0[h] + b_hh0[h];
        #pragma unroll 8
        for (int c = 0; c < 32; ++c) s += ctx[b][c] * w_ih0[h * 33 + 1 + c];
        cbase[b * HID + h] = s;
    }
}

// ---------------------------------------------------------------------------
// 2) Recurrence: h_t = tanh(pre_t + h_{t-1} @ w_hh.T)
//    2 workgroups (16 batch rows each), 8 waves, wave w owns columns
//    [32w, 32w+32). w_hh.T fragments (bf16) held in 128 VGPRs per lane for
//    the whole 8192-step loop. h double-buffered in LDS in A-fragment
//    layout; one workgroup barrier per step. Mode is a template parameter
//    so the t-loop body is branch-free; pre/audio loads are issued before
//    the WMMA chain so global latency hides behind matrix work; the A
//    fragment is software-pipelined one kt ahead to overlap LDS latency.
// ---------------------------------------------------------------------------
template <int USE_AUDIO>
__global__ __launch_bounds__(256)
void rec_kernel(const float* __restrict__ w_hh,     // (256,256) f32
                const float* __restrict__ pre,      // (B*T,256) f32 (mode 0)
                const float* __restrict__ x,        // raw input  (mode 1)
                const float* __restrict__ cbase,    // (32,256)   (mode 1)
                const float* __restrict__ w_ih0,    // (256,33)   (mode 1)
                __bf16* __restrict__ outb)          // (B*T,256) bf16
{
    __shared__ __bf16 hbuf[2][16][HPAD];

    const int tid  = threadIdx.x;
    const int wave = tid >> 5;
    const int lane = tid & 31;
    const int lm   = lane & 15;     // row (A) or col (B/C) within tile
    const int lg   = lane >> 4;     // half-select
    const int b0   = blockIdx.x * 16;

    // ---- preload B = w_hh.T fragments as bf16 into registers -------------
    // B 32x16 bf16 layout: lane<16 -> N=lane, K=kt*32+0..15 ; lane>=16 -> K=+16
    v16bf bfrag[2][8];
    #pragma unroll
    for (int ntl = 0; ntl < 2; ++ntl) {
        int n = wave * 32 + ntl * 16 + lm;
        #pragma unroll
        for (int kt = 0; kt < 8; ++kt) {
            const float* src = w_hh + n * HID + kt * 32 + lg * 16;
            v16bf tmp;
            #pragma unroll
            for (int e = 0; e < 16; ++e) tmp[e] = (__bf16)src[e];
            bfrag[ntl][kt] = tmp;
        }
    }

    // ---- audio-mode constants -------------------------------------------
    float w0[2] = {0.f, 0.f};
    float cb[2][8];
    if (USE_AUDIO) {
        #pragma unroll
        for (int ntl = 0; ntl < 2; ++ntl) {
            int n = wave * 32 + ntl * 16 + lm;
            w0[ntl] = w_ih0[n * 33];
            #pragma unroll
            for (int r = 0; r < 8; ++r)
                cb[ntl][r] = cbase[(b0 + r + 8 * lg) * HID + n];
        }
    }

    // ---- h0 = 0 ----------------------------------------------------------
    for (int i = tid; i < 2 * 16 * HPAD; i += 256)
        (&hbuf[0][0][0])[i] = (__bf16)0.0f;
    __syncthreads();

    int p = 0;
    for (int t = 0; t < TSEQ; ++t) {
        // ---- (1) issue independent global loads first (hide behind WMMA) --
        float pv[2][8];
        if (USE_AUDIO) {
            float au[8];
            #pragma unroll
            for (int r = 0; r < 8; ++r)
                au[r] = x[((long)(b0 + r + 8 * lg) * TSEQ + t) * FIN];
            #pragma unroll
            for (int ntl = 0; ntl < 2; ++ntl)
                #pragma unroll
                for (int r = 0; r < 8; ++r)
                    pv[ntl][r] = fmaf(au[r], w0[ntl], cb[ntl][r]);
        } else {
            #pragma unroll
            for (int ntl = 0; ntl < 2; ++ntl) {
                int n = wave * 32 + ntl * 16 + lm;
                #pragma unroll
                for (int r = 0; r < 8; ++r)
                    pv[ntl][r] = pre[((long)(b0 + r + 8 * lg) * TSEQ + t) * HID + n];
            }
            // branchless near-prefetch of next step's tile (speculative; the
            // final overshoot lands in ws scratch, harmless)
            __builtin_prefetch(&pre[((long)(b0 + lm) * TSEQ + (t + 1)) * HID + wave * 32], 0, 3);
        }

        // ---- (2) WMMA chain, A fragment pipelined one kt ahead ------------
        const __bf16* hrow = &hbuf[p][lm][lg * 16];
        v8f acc[2] = {};
        v16bf a_cur, a_nxt;
        ((v8bf*)&a_cur)[0] = *(const v8bf*)(hrow);
        ((v8bf*)&a_cur)[1] = *(const v8bf*)(hrow + 8);
        #pragma unroll
        for (int kt = 0; kt < 8; ++kt) {
            if (kt < 7) {
                const __bf16* ar = hrow + (kt + 1) * 32;
                ((v8bf*)&a_nxt)[0] = *(const v8bf*)(ar);
                ((v8bf*)&a_nxt)[1] = *(const v8bf*)(ar + 8);
            }
            acc[0] = wmma_bf16(a_cur, bfrag[0][kt], acc[0]);
            acc[1] = wmma_bf16(a_cur, bfrag[1][kt], acc[1]);
            a_cur = a_nxt;
        }

        // ---- (3) tanh(acc + pre) -> new h (bf16) into hbuf[p^1] -----------
        #pragma unroll
        for (int ntl = 0; ntl < 2; ++ntl) {
            int n = wave * 32 + ntl * 16 + lm;
            #pragma unroll
            for (int r = 0; r < 8; ++r) {
                int m = r + 8 * lg;
                hbuf[p ^ 1][m][n] = (__bf16)fast_tanh(acc[ntl][r] + pv[ntl][r]);
            }
        }
        __syncthreads();

        // ---- (4) cooperative coalesced store of h_t (512B per (b,t) row) --
        {
            int m = tid >> 4, seg = tid & 15;
            const v8bf* s = (const v8bf*)&hbuf[p ^ 1][m][seg * 16];
            v8bf* d = (v8bf*)(outb + ((long)(b0 + m) * TSEQ + t) * HID + seg * 16);
            d[0] = s[0];
            d[1] = s[1];
        }
        p ^= 1;
    }
}

// ---------------------------------------------------------------------------
// 3) Input projection GEMM: pre = A(bf16, Mx256) @ w_ih.T + (b_ih + b_hh)
//    256 thr / 8 waves per WG, 128 M-rows per WG, wave w -> N columns [32w,32w+32)
// ---------------------------------------------------------------------------
__global__ __launch_bounds__(256)
void proj_kernel(const __bf16* __restrict__ a,     // (M,256) bf16
                 const float* __restrict__ w,      // (256,256) w_ih
                 const float* __restrict__ bi, const float* __restrict__ bh,
                 float* __restrict__ pre)          // (M,256) f32
{
    const int tid  = threadIdx.x;
    const int wave = tid >> 5;
    const int lane = tid & 31;
    const int lm   = lane & 15;
    const int lg   = lane >> 4;
    const long mbase = (long)blockIdx.x * 128;

    v16bf bfrag[2][8];
    float bias[2];
    #pragma unroll
    for (int ntl = 0; ntl < 2; ++ntl) {
        int n = wave * 32 + ntl * 16 + lm;
        bias[ntl] = bi[n] + bh[n];
        #pragma unroll
        for (int kt = 0; kt < 8; ++kt) {
            const float* src = w + n * HID + kt * 32 + lg * 16;
            v16bf tmp;
            #pragma unroll
            for (int e = 0; e < 16; ++e) tmp[e] = (__bf16)src[e];
            bfrag[ntl][kt] = tmp;
        }
    }

    for (int mt = 0; mt < 8; ++mt) {
        long mrow = mbase + mt * 16 + lm;
        v8f acc[2] = {};
        #pragma unroll
        for (int kt = 0; kt < 8; ++kt) {
            v16bf afr;
            const __bf16* ar = a + mrow * HID + kt * 32 + lg * 16;
            ((v8bf*)&afr)[0] = *(const v8bf*)ar;
            ((v8bf*)&afr)[1] = *(const v8bf*)(ar + 8);
            acc[0] = wmma_bf16(afr, bfrag[0][kt], acc[0]);
            acc[1] = wmma_bf16(afr, bfrag[1][kt], acc[1]);
        }
        #pragma unroll
        for (int ntl = 0; ntl < 2; ++ntl) {
            int n = wave * 32 + ntl * 16 + lm;
            #pragma unroll
            for (int r = 0; r < 8; ++r)
                pre[(mbase + mt * 16 + r + 8 * lg) * HID + n] = acc[ntl][r] + bias[ntl];
        }
    }
}

// ---------------------------------------------------------------------------
// 4) Head: y = gelu(out @ q(Wh1).T + bh1) @ q(Wh2).T + bh2
//    Wave w owns N-tile w (N=128). GELU result staged in LDS, then 128
//    threads each reduce one row against quantized Wh2.
// ---------------------------------------------------------------------------
__global__ __launch_bounds__(256)
void head_kernel(const __bf16* __restrict__ a,     // (M,256) bf16
                 const float* __restrict__ Wh1, const float* __restrict__ bh1,
                 const float* __restrict__ Wh2, const float* __restrict__ bh2,
                 float* __restrict__ out)          // (M,)
{
    __shared__ float g[128][132];
    __shared__ float wq2[128];

    const int tid  = threadIdx.x;
    const int wave = tid >> 5;
    const int lane = tid & 31;
    const int lm   = lane & 15;
    const int lg   = lane >> 4;
    const long mbase = (long)blockIdx.x * 128;

    v16bf bfrag[8];
    const int n = wave * 16 + lm;
    const float bias = bh1[n];
    #pragma unroll
    for (int kt = 0; kt < 8; ++kt) {
        const float* src = Wh1 + n * HID + kt * 32 + lg * 16;
        v16bf tmp;
        #pragma unroll
        for (int e = 0; e < 16; ++e) tmp[e] = (__bf16)fq(src[e]);
        bfrag[kt] = tmp;
    }
    if (tid < 128) wq2[tid] = fq(Wh2[tid]);

    for (int mt = 0; mt < 8; ++mt) {
        long mrow = mbase + mt * 16 + lm;
        v8f acc = {};
        #pragma unroll
        for (int kt = 0; kt < 8; ++kt) {
            v16bf afr;
            const __bf16* ar = a + mrow * HID + kt * 32 + lg * 16;
            ((v8bf*)&afr)[0] = *(const v8bf*)ar;
            ((v8bf*)&afr)[1] = *(const v8bf*)(ar + 8);
            acc = wmma_bf16(afr, bfrag[kt], acc);
        }
        #pragma unroll
        for (int r = 0; r < 8; ++r) {
            float v = acc[r] + bias;
            float gel = 0.5f * v * (1.0f + erff(v * 0.70710678118654752f));
            g[mt * 16 + r + 8 * lg][n] = gel;
        }
    }
    __syncthreads();

    if (tid < 128) {
        float s = bh2[0];
        #pragma unroll 16
        for (int k = 0; k < 128; ++k) s += g[tid][k] * wq2[k];
        out[mbase + tid] = s;
    }
}

// ---------------------------------------------------------------------------
extern "C" void kernel_launch(void* const* d_in, const int* in_sizes, int n_in,
                              void* d_out, int out_size, void* d_ws, size_t ws_size,
                              hipStream_t stream)
{
    (void)in_sizes; (void)n_in; (void)out_size; (void)ws_size;
    const float* x     = (const float*)d_in[0];
    const float* W1    = (const float*)d_in[1];
    const float* b1    = (const float*)d_in[2];
    const float* W2    = (const float*)d_in[3];
    const float* b2    = (const float*)d_in[4];
    const float* w_ih0 = (const float*)d_in[5];
    const float* w_hh0 = (const float*)d_in[6];
    const float* b_ih0 = (const float*)d_in[7];
    const float* b_hh0 = (const float*)d_in[8];
    const float* w_ih1 = (const float*)d_in[9];
    const float* w_hh1 = (const float*)d_in[10];
    const float* b_ih1 = (const float*)d_in[11];
    const float* b_hh1 = (const float*)d_in[12];
    const float* w_ih2 = (const float*)d_in[13];
    const float* w_hh2 = (const float*)d_in[14];
    const float* b_ih2 = (const float*)d_in[15];
    const float* b_hh2 = (const float*)d_in[16];
    const float* Wh1   = (const float*)d_in[17];
    const float* bh1   = (const float*)d_in[18];
    const float* Wh2   = (const float*)d_in[19];
    const float* bh2   = (const float*)d_in[20];

    char* ws = (char*)d_ws;
    const size_t OUT_BYTES = (size_t)BATCH * TSEQ * HID * 2;   // 134 MB bf16
    const size_t PRE_BYTES = (size_t)BATCH * TSEQ * HID * 4;   // 268 MB f32
    __bf16* out_buf = (__bf16*)ws;
    float*  pre_buf = (float*)(ws + OUT_BYTES);
    float*  cbase   = (float*)(ws + OUT_BYTES + PRE_BYTES);

    const int M_BLOCKS = (BATCH * TSEQ) / 128;   // 2048

    ctx_kernel<<<1, 256, 0, stream>>>(x, W1, b1, W2, b2, w_ih0, b_ih0, b_hh0, cbase);

    // layer 0: fused audio projection + recurrence
    rec_kernel<1><<<2, 256, 0, stream>>>(w_hh0, nullptr, x, cbase, w_ih0, out_buf);

    // layer 1
    proj_kernel<<<M_BLOCKS, 256, 0, stream>>>(out_buf, w_ih1, b_ih1, b_hh1, pre_buf);
    rec_kernel<0><<<2, 256, 0, stream>>>(w_hh1, pre_buf, nullptr, nullptr, nullptr, out_buf);

    // layer 2
    proj_kernel<<<M_BLOCKS, 256, 0, stream>>>(out_buf, w_ih2, b_ih2, b_hh2, pre_buf);
    rec_kernel<0><<<2, 256, 0, stream>>>(w_hh2, pre_buf, nullptr, nullptr, nullptr, out_buf);

    // head
    head_kernel<<<M_BLOCKS, 256, 0, stream>>>(out_buf, Wh1, bh1, Wh2, bh2, (float*)d_out);
}
